// CLSTMCell_43224550867775
// MI455X (gfx1250) — matrix operand order repaired and checked by reading, not compiled, not run on hardware
//
#include <hip/hip_runtime.h>
#include <hip/hip_bf16.h>
#include <stdint.h>

// Complex LSTM cell, fused for MI455X (gfx1250, wave32).
// Compute-bound (275 GFLOP vs ~200MB => ~1375 FLOP/B): bf16x3-split
// V_WMMA_F32_16X16X32_BF16 (fp32-accurate, ~2.7x native f32 WMMA rate).
// Fast path: fp32->bf16(hi,lo) pre-split into workspace once; X tiles staged
// into double-buffered LDS with global_load_async_to_lds_b128 (ASYNCcnt)
// overlapped with WMMA compute; weights stream from L2 (64MB, 192MB L2).

#define B_DIM  4096
#define IN_DIM 1024
#define H_DIM  1024
#define KCAT   4096

typedef __bf16 v16bf __attribute__((ext_vector_type(16)));
typedef float  v8f   __attribute__((ext_vector_type(8)));

union BFrag {
    v16bf v;
    unsigned short u[16];
    uint4 q[2];
};

__device__ __forceinline__ unsigned short f2bf_rne(float f) {
    unsigned int u = __float_as_uint(f);
    u += 0x7FFFu + ((u >> 16) & 1u);          // round-to-nearest-even
    return (unsigned short)(u >> 16);
}
__device__ __forceinline__ float bf2f(unsigned short s) {
    return __uint_as_float(((unsigned int)s) << 16);
}
__device__ __forceinline__ float sigmoidf_(float x) {
    return 1.0f / (1.0f + __expf(-x));
}
__device__ __forceinline__ void split4(float4 v, ushort4& hs, ushort4& ls) {
    hs.x = f2bf_rne(v.x); ls.x = f2bf_rne(v.x - bf2f(hs.x));
    hs.y = f2bf_rne(v.y); ls.y = f2bf_rne(v.y - bf2f(hs.y));
    hs.z = f2bf_rne(v.z); ls.z = f2bf_rne(v.z - bf2f(hs.z));
    hs.w = f2bf_rne(v.w); ls.w = f2bf_rne(v.w - bf2f(hs.w));
}
__device__ __forceinline__ void wait_async0() {
#if __has_builtin(__builtin_amdgcn_s_wait_asynccnt)
    __builtin_amdgcn_s_wait_asynccnt(0);
#else
    asm volatile("s_wait_asynccnt 0x0" ::: "memory");
#endif
}
__device__ __forceinline__ void async_copy_b128(unsigned lds_off, const void* g) {
    asm volatile("global_load_async_to_lds_b128 %0, %1, off"
                 :: "v"(lds_off), "v"((unsigned long long)(uintptr_t)g) : "memory");
}

// ---------------- prep: split X (cat layout [B][4096]) into bf16 hi/lo ----
__global__ __launch_bounds__(256) void split_x_kernel(
    const float* __restrict__ inp, const float* __restrict__ h_x,
    unsigned short* __restrict__ Xhi, unsigned short* __restrict__ Xlo)
{
    const size_t e = ((size_t)blockIdx.x * 256 + threadIdx.x) * 4;  // 16M elems
    const int b = (int)(e >> 12);
    const int k = (int)(e & 4095);
    const int seg = k >> 10, d = k & 1023;
    const float* src = (seg < 2) ? inp : h_x;
    const float4 v = *(const float4*)(src + (size_t)b * 2048 + (seg & 1) * 1024 + d);
    ushort4 hs, ls;
    split4(v, hs, ls);
    *(ushort4*)(Xhi + e) = hs;
    *(ushort4*)(Xlo + e) = ls;
}

// ---------------- prep: split weights [Uw_r|Uw_i|Ww_r|Ww_i] into bf16 hi/lo
__global__ __launch_bounds__(256) void split_w_kernel(
    const float* __restrict__ Uw_r, const float* __restrict__ Uw_i,
    const float* __restrict__ Ww_r, const float* __restrict__ Ww_i,
    unsigned short* __restrict__ Whi, unsigned short* __restrict__ Wlo)
{
    const size_t e = ((size_t)blockIdx.x * 256 + threadIdx.x) * 4;  // 16M elems
    const int arr = (int)(e >> 22);                                 // block-uniform
    const size_t idx = e & 0x3FFFFFu;
    const float* src = (arr == 0) ? Uw_r : (arr == 1) ? Uw_i : (arr == 2) ? Ww_r : Ww_i;
    const float4 v = *(const float4*)(src + idx);
    ushort4 hs, ls;
    split4(v, hs, ls);
    *(ushort4*)(Whi + e) = hs;
    *(ushort4*)(Wlo + e) = ls;
}

// ---------------- fused GEMM (bf16x3 WMMA) + gating ----------------
// LDS arena layout (40960B, all phases barrier-separated):
//   buf b in [0,1]: Ahi at b*20480, Alo at b*20480+10240 (bytes); rows padded
//   to 40 elems (80B) -> conflict-free, 16B-aligned b128 fragments.
//   epilogue: zbuf (8*16*64 f32 = 32768B) aliases the arena.
template <bool PRESPLIT>
__global__ __launch_bounds__(1024)
void clstm_fused_kernel(const float* __restrict__ inp,
                        const float* __restrict__ h_x,
                        const float* __restrict__ c_x,
                        const float* __restrict__ Uw_r, const float* __restrict__ Uw_i,
                        const float* __restrict__ Ub_r, const float* __restrict__ Ub_i,
                        const float* __restrict__ Ww_r, const float* __restrict__ Ww_i,
                        const float* __restrict__ Wb_r, const float* __restrict__ Wb_i,
                        const unsigned short* __restrict__ Xhi,
                        const unsigned short* __restrict__ Xlo,
                        const unsigned short* __restrict__ Whi,
                        const unsigned short* __restrict__ Wlo,
                        float* __restrict__ out)
{
    __shared__ alignas(16) unsigned char smem[40960];
    unsigned short* S = (unsigned short*)smem;      // ushort view of arena
    float* zbuf = (float*)smem;                     // epilogue alias

    const int tid   = threadIdx.x;
    const int lane  = tid & 31;
    const int wave  = tid >> 5;
    const int c     = wave & 7;                // 0-3 real gates, 4-7 imag gates
    const int nsub  = wave >> 3;               // 0..3 : 16-wide N slice
    const int part  = c >> 2;
    const int g     = c & 3;
    const int lo16  = lane & 15;
    const int hi16  = lane >> 4;

    const int mbase = blockIdx.y * 128;
    const int n0    = blockIdx.x * 64;

    const int o    = n0 + nsub * 16 + lo16;    // lane's output column (B-frag N)
    const int kb   = hi16 * 16;                // B-frag K split: 0..15 / 16..31
    const int koff = hi16 * 16;                // A-frag K split: 0..15 / 16..31

    // staging indices
    const int arow  = tid >> 3;                // fallback: fp32 row, 0..127
    const int acg   = (tid & 7) * 4;
    const int spart = tid >> 9;                // presplit: 0 = hi, 1 = lo
    const int srow  = (tid & 511) >> 2;        // 0..127
    const int sc16  = (tid & 3) * 8;           // bf16 col offset 0,8,16,24

    v8f acc[8];
    #pragma unroll
    for (int m = 0; m < 8; ++m)
        #pragma unroll
        for (int j = 0; j < 8; ++j) acc[m][j] = 0.0f;

    // double-buffer state (PRESPLIT): running global src ptr, LDS dst offsets
    const unsigned short* gptr = nullptr;
    unsigned lbuf0 = 0, lbuf1 = 0;
    if constexpr (PRESPLIT) {
        gptr = (spart ? Xlo : Xhi) + (size_t)(mbase + srow) * KCAT + sc16;
        unsigned short* l0 = S + spart * 5120 + srow * 40 + sc16;
        lbuf0 = (unsigned)(uintptr_t)l0;
        lbuf1 = lbuf0 + 20480u;
        // prologue: fill buffer 0 with K-tile 0
        async_copy_b128(lbuf0, gptr);
        gptr += 32;
        wait_async0();
        __syncthreads();
    }
    int cur = 0;

    #pragma unroll 1
    for (int seg = 0; seg < 4; ++seg) {        // K segs: xr, xi, hr, hi
        // weight selection: part0 -> [Ur,+][Ui,-][Wr,+][Wi,-] ; part1 -> [Ui][Ur][Wi][Wr]
        const int  warr = part ? (seg ^ 1) : seg;
        const bool neg  = (!part) && (seg & 1);
        const unsigned smask = neg ? 0x80008000u : 0u;

        const float* xsrc = (seg < 2) ? inp : h_x;
        const int    xoff = (seg & 1) * 1024;

        const float* Wseg32;
        {
            const float* tbl0 = part ? Uw_i : Uw_r;
            const float* tbl1 = part ? Uw_r : Uw_i;
            const float* tbl2 = part ? Ww_i : Ww_r;
            const float* tbl3 = part ? Ww_r : Ww_i;
            Wseg32 = (seg == 0) ? tbl0 : (seg == 1) ? tbl1 : (seg == 2) ? tbl2 : tbl3;
        }
        const float sgn = neg ? -1.0f : 1.0f;
        const size_t wrowidx = (size_t)(g * H_DIM + o) * IN_DIM;
        const float* wrow32 = Wseg32 + wrowidx;
        const unsigned short* wrow_hi = Whi + ((size_t)warr << 22) + wrowidx;
        const unsigned short* wrow_lo = Wlo + ((size_t)warr << 22) + wrowidx;

        #pragma unroll 1
        for (int kt = 0; kt < 32; ++kt) {
            const int d0 = kt * 32;

            if constexpr (PRESPLIT) {
                // ---- issue async copy of NEXT K-tile into the idle buffer ----
                if (!(seg == 3 && kt == 31)) {
                    async_copy_b128(cur ? lbuf0 : lbuf1, gptr);
                    gptr += 32;
                }
            } else {
                // ---- stage X tile with in-kernel fp32 -> bf16 hi/lo split ----
                __syncthreads();
                const float4 xv = *(const float4*)(xsrc + (size_t)(mbase + arow) * 2048
                                                   + xoff + d0 + acg);
                ushort4 hs, ls;
                split4(xv, hs, ls);
                const int base = arow * 40 + acg;
                *(ushort4*)&S[base]        = hs;
                *(ushort4*)&S[5120 + base] = ls;
                __syncthreads();
            }

            // ---- B fragment: 16 contiguous K of this lane's weight row ----
            BFrag bhi, blo;
            if constexpr (PRESPLIT) {
                const unsigned short* wp = wrow_hi + d0 + kb;
                const unsigned short* wq = wrow_lo + d0 + kb;
                bhi.q[0] = *(const uint4*)(wp);
                bhi.q[1] = *(const uint4*)(wp + 8);
                blo.q[0] = *(const uint4*)(wq);
                blo.q[1] = *(const uint4*)(wq + 8);
                #pragma unroll
                for (int q = 0; q < 2; ++q) {     // sign = xor of bf16 sign bits
                    bhi.q[q].x ^= smask; bhi.q[q].y ^= smask;
                    bhi.q[q].z ^= smask; bhi.q[q].w ^= smask;
                    blo.q[q].x ^= smask; blo.q[q].y ^= smask;
                    blo.q[q].z ^= smask; blo.q[q].w ^= smask;
                }
                __builtin_prefetch(wp + 32, 0, 0);
                __builtin_prefetch(wq + 32, 0, 0);
            } else {
                const float* wp = wrow32 + d0 + kb;
                #pragma unroll
                for (int q = 0; q < 4; ++q) {
                    const float4 wv = *(const float4*)(wp + q * 4);
                    float f0 = wv.x * sgn, f1 = wv.y * sgn, f2 = wv.z * sgn, f3 = wv.w * sgn;
                    bhi.u[q*4+0] = f2bf_rne(f0); blo.u[q*4+0] = f2bf_rne(f0 - bf2f(bhi.u[q*4+0]));
                    bhi.u[q*4+1] = f2bf_rne(f1); blo.u[q*4+1] = f2bf_rne(f1 - bf2f(bhi.u[q*4+1]));
                    bhi.u[q*4+2] = f2bf_rne(f2); blo.u[q*4+2] = f2bf_rne(f2 - bf2f(bhi.u[q*4+2]));
                    bhi.u[q*4+3] = f2bf_rne(f3); blo.u[q*4+3] = f2bf_rne(f3 - bf2f(bhi.u[q*4+3]));
                }
                __builtin_prefetch(wp + 32, 0, 0);
            }

            // ---- 8 M-subtiles x bf16x3 WMMA on the current buffer ----
            const int abuf = cur * 10240;         // element offset of active buffer
            #pragma unroll
            for (int m = 0; m < 8; ++m) {
                BFrag ah, al;
                const int abase = abuf + (m * 16 + lo16) * 40 + koff;
                ah.q[0] = *(const uint4*)&S[abase];
                ah.q[1] = *(const uint4*)&S[abase + 8];
                al.q[0] = *(const uint4*)&S[abase + 5120];
                al.q[1] = *(const uint4*)&S[abase + 5120 + 8];
                acc[m] = __builtin_amdgcn_wmma_f32_16x16x32_bf16(
                            false, ah.v, false, bhi.v, (short)0, acc[m], false, false);
                acc[m] = __builtin_amdgcn_wmma_f32_16x16x32_bf16(
                            false, ah.v, false, blo.v, (short)0, acc[m], false, false);
                acc[m] = __builtin_amdgcn_wmma_f32_16x16x32_bf16(
                            false, al.v, false, bhi.v, (short)0, acc[m], false, false);
            }

            if constexpr (PRESPLIT) {
                wait_async0();                    // my next-tile copy has landed
                __syncthreads();                  // everyone's has landed
                cur ^= 1;
            }
        }
    }

    // ---- bias (depends only on this lane's output column o) ----
    {
        const float* b1 = part ? Ub_i : Ub_r;
        const float* b2 = part ? Wb_i : Wb_r;
        const float bias = b1[g * H_DIM + o] + b2[g * H_DIM + o];
        #pragma unroll
        for (int m = 0; m < 8; ++m)
            #pragma unroll
            for (int j = 0; j < 8; ++j) acc[m][j] += bias;
    }

    // ---- gate exchange + elementwise epilogue, 16 batch rows at a time ----
    const int colw = nsub * 16 + lo16;
    const int ml   = tid >> 6;                  // 0..15
    const int col  = tid & 63;                  // 0..63
    #pragma unroll
    for (int m = 0; m < 8; ++m) {
        __syncthreads();
        #pragma unroll
        for (int j = 0; j < 8; ++j) {
            const int r = hi16 * 8 + j;         // C/D layout: VGPR j -> M=j (+8 hi half)
            zbuf[(c * 16 + r) * 64 + colw] = acc[m][j];
        }
        __syncthreads();

        const int   b  = mbase + m * 16 + ml;
        const int   oo = n0 + col;
        const float zrf = zbuf[(0*16 + ml)*64 + col], zif = zbuf[(4*16 + ml)*64 + col];
        const float zri = zbuf[(1*16 + ml)*64 + col], zii = zbuf[(5*16 + ml)*64 + col];
        const float zra = zbuf[(2*16 + ml)*64 + col], zia = zbuf[(6*16 + ml)*64 + col];
        const float zro = zbuf[(3*16 + ml)*64 + col], zio = zbuf[(7*16 + ml)*64 + col];

        const float fr  = sigmoidf_(zrf), fi  = sigmoidf_(zif);
        const float ir  = sigmoidf_(zri), ii_ = sigmoidf_(zii);
        const float ar  = tanhf(zra),     ai  = tanhf(zia);
        const float orr = sigmoidf_(zro), oi  = sigmoidf_(zio);

        const float cr = c_x[(size_t)b * 2048 + oo];
        const float ci = c_x[(size_t)b * 2048 + 1024 + oo];

        const float ctr = (cr * fr - ci * fi) + (ar * ir - ai * ii_);
        const float cti = (cr * fi + ci * fr) + (ar * ii_ + ai * ir);
        const float thr = tanhf(ctr), thi = tanhf(cti);
        const float htr = orr * thr - oi * thi;
        const float hti = orr * thi + oi * thr;

        out[(size_t)b * 2048 + oo]        = htr;
        out[(size_t)b * 2048 + 1024 + oo] = hti;
        out[(size_t)B_DIM * 2048 + (size_t)b * 2048 + oo]        = ctr;
        out[(size_t)B_DIM * 2048 + (size_t)b * 2048 + 1024 + oo] = cti;
    }
}

extern "C" void kernel_launch(void* const* d_in, const int* in_sizes, int n_in,
                              void* d_out, int out_size, void* d_ws, size_t ws_size,
                              hipStream_t stream) {
    const float* inp  = (const float*)d_in[0];
    const float* h_x  = (const float*)d_in[1];
    const float* c_x  = (const float*)d_in[2];
    const float* Uw_r = (const float*)d_in[3];
    const float* Uw_i = (const float*)d_in[4];
    const float* Ub_r = (const float*)d_in[5];
    const float* Ub_i = (const float*)d_in[6];
    const float* Ww_r = (const float*)d_in[7];
    const float* Ww_i = (const float*)d_in[8];
    const float* Wb_r = (const float*)d_in[9];
    const float* Wb_i = (const float*)d_in[10];
    float* out = (float*)d_out;

    const dim3 grid(H_DIM / 64, B_DIM / 128);  // 16 x 32
    const dim3 block(1024);                    // 32 wave32s

    const size_t need = (size_t)128 * 1024 * 1024;  // Xhi/Xlo/Whi/Wlo, 32MB each
    if (d_ws != nullptr && ws_size >= need) {
        unsigned short* Xhi = (unsigned short*)d_ws;
        unsigned short* Xlo = Xhi + (size_t)16 * 1024 * 1024;
        unsigned short* Whi = Xlo + (size_t)16 * 1024 * 1024;
        unsigned short* Wlo = Whi + (size_t)16 * 1024 * 1024;
        hipLaunchKernelGGL(split_x_kernel, dim3(16384), dim3(256), 0, stream,
                           inp, h_x, Xhi, Xlo);
        hipLaunchKernelGGL(split_w_kernel, dim3(16384), dim3(256), 0, stream,
                           Uw_r, Uw_i, Ww_r, Ww_i, Whi, Wlo);
        hipLaunchKernelGGL((clstm_fused_kernel<true>), grid, block, 0, stream,
                           inp, h_x, c_x, Uw_r, Uw_i, Ub_r, Ub_i,
                           Ww_r, Ww_i, Wb_r, Wb_i, Xhi, Xlo, Whi, Wlo, out);
    } else {
        hipLaunchKernelGGL((clstm_fused_kernel<false>), grid, block, 0, stream,
                           inp, h_x, c_x, Uw_r, Uw_i, Ub_r, Ub_i,
                           Ww_r, Ww_i, Wb_r, Wb_i,
                           (const unsigned short*)nullptr, (const unsigned short*)nullptr,
                           (const unsigned short*)nullptr, (const unsigned short*)nullptr,
                           out);
    }
}
